// ScratchGMIC_83348135346349
// MI455X (gfx1250) — matrix operand order repaired and checked by reading, not compiled, not run on hardware
//
#include <hip/hip_runtime.h>
#include <hip/hip_bf16.h>
#include <math.h>

typedef __attribute__((ext_vector_type(16))) _Float16 v16h;
typedef __attribute__((ext_vector_type(8)))  float    v8f;

// TDM descriptor group types (6-arg builtin on this toolchain)
typedef __attribute__((ext_vector_type(4))) unsigned int tdu4;
typedef __attribute__((ext_vector_type(8))) int          tdi8;
typedef __attribute__((ext_vector_type(4))) int          tdi4;

// async LDS-DMA payload type: b128 = 4 x i32, AS1 source / AS3 destination
typedef __attribute__((ext_vector_type(4))) int          asv4i;
typedef __attribute__((address_space(1))) asv4i          g_v4i;
typedef __attribute__((address_space(3))) asv4i          l_v4i;

#define IMG_H 2944
#define IMG_W 1920
#define CAMH 46
#define CAMW 30
#define POOLH 43
#define POOLW 27

// Feature gates: must be evaluated per compilation pass (host pass lacks the
// amdgcn builtins, so kernels keep a portable fallback path in-body).
#if defined(__AMDGCN__) && __has_builtin(__builtin_amdgcn_global_load_async_to_lds_b128)
#define USE_ASYNC_LDS 1
#else
#define USE_ASYNC_LDS 0
#endif
#if defined(__AMDGCN__) && __has_builtin(__builtin_amdgcn_tensor_load_to_lds)
#define USE_TDM 1
#else
#define USE_TDM 0
#endif

// ---------------------------------------------------------------------------
// Weight packing: (O, I, kh, kw) fp32 -> [tap][O_pad32][I_pad32] f16, zero pad
// ---------------------------------------------------------------------------
__global__ void k_pack_w(const float* __restrict__ w, _Float16* __restrict__ wpk,
                         int Cout, int Cin, int ks, int Cout_p, int Cin_p) {
    size_t total = (size_t)ks * ks * Cout_p * Cin_p;
    size_t i = (size_t)blockIdx.x * 256 + threadIdx.x;
    if (i >= total) return;
    int c   = (int)(i % Cin_p);
    size_t t = i / Cin_p;
    int o   = (int)(t % Cout_p);
    int tap = (int)(t / Cout_p);
    int kh = tap / ks, kw = tap % ks;
    float v = 0.f;
    if (o < Cout && c < Cin)
        v = w[(((size_t)o * Cin + c) * ks + kh) * ks + kw];
    wpk[i] = (_Float16)v;
}

// ---------------------------------------------------------------------------
// Implicit-GEMM conv, NCHW, WMMA f32_16x16x32_f16.
// Wave tile: 16 (Cout) x 16 (positions). Block = 256 thr = 8 waves = 32x64.
// Per (tap, k0) the 32x32 f16 weight chunk is staged once into LDS (async
// LDS-DMA when available) and shared by all 8 waves; A-fragments then come
// from ds_load. Epilogue: optional BN, optional residual add, optional ReLU.
// ---------------------------------------------------------------------------
__global__ void __launch_bounds__(256)
k_conv_wmma(const float* __restrict__ x, const _Float16* __restrict__ wpk,
            float* __restrict__ y, const float* __restrict__ res,
            const float* __restrict__ bng, const float* __restrict__ bnb,
            const float* __restrict__ bnm, const float* __restrict__ bnv,
            int N, int Cin, int Hin, int Win, int Cout, int Hout, int Wout,
            int ks, int stride, int pad, int Cin_p, int Cout_p, int act) {
    (void)N;
    __shared__ __align__(16) _Float16 swt[32 * 32];     // one 32x32 weight chunk

    const int HWo  = Hout * Wout;
    const int lane = threadIdx.x & 31;
    const int wv   = threadIdx.x >> 5;
    const int wm   = wv >> 2;          // 0..1 : Cout 16-row group
    const int wn   = wv & 3;           // 0..3 : position 16-col group
    const int coTiles = Cout_p >> 5;
    const int n   = blockIdx.y / coTiles;
    const int cob = (blockIdx.y % coTiles) * 32;
    const int co0 = cob + wm * 16;
    const int p0  = blockIdx.x * 64 + wn * 16;
    const int l15 = lane & 15;
    const int khf = lane >> 4;

    const int aRow = wm * 16 + l15;    // row within the staged 32-row chunk
    const int p  = p0 + l15;           // B-fragment column (output position)
    const bool pv = (p < HWo);
    const int pc  = pv ? p : 0;
    const int oy  = pc / Wout, ox = pc % Wout;
    const size_t cstr = (size_t)Hin * Win;
    const float* xb = x + (size_t)n * Cin * cstr;

    const int t   = threadIdx.x;
    const int srow = t >> 2, sseg = t & 3;              // staging: 32 rows x 4 segs

    v8f acc = {};
    for (int kh = 0; kh < ks; ++kh) {
        const int iy = oy * stride - pad + kh;
        for (int kw = 0; kw < ks; ++kw) {
            const int ix = ox * stride - pad + kw;
            const bool inb = pv && iy >= 0 && iy < Hin && ix >= 0 && ix < Win;
            const size_t tapoff = (size_t)(kh * ks + kw) * Cout_p * Cin_p;
            const float* xt = xb + (size_t)iy * Win + ix;
            if (inb) __builtin_prefetch(xt, 0, 1);      // global_prefetch_b8
            for (int k0 = 0; k0 < Cin_p; k0 += 32) {
                // ---- stage 32x32 f16 weight chunk into LDS (2 KB) ----
                if (t < 128) {
                    const _Float16* src =
                        wpk + tapoff + (size_t)(cob + srow) * Cin_p + k0 + sseg * 8;
#if USE_ASYNC_LDS
                    __builtin_amdgcn_global_load_async_to_lds_b128(
                        (g_v4i*)src,
                        (l_v4i*)&swt[srow * 32 + sseg * 8],
                        0, 0);
#else
                    *(uint4*)&swt[srow * 32 + sseg * 8] = *(const uint4*)src;
#endif
                }
#if USE_ASYNC_LDS
#if __has_builtin(__builtin_amdgcn_s_wait_asynccnt)
                __builtin_amdgcn_s_wait_asynccnt(0);
#else
                asm volatile("s_wait_asynccnt 0x0" ::: "memory");
#endif
#endif
                __syncthreads();

                v16h a, b;
#pragma unroll
                for (int e = 0; e < 16; ++e) {          // A 16x32 f16 layout (ISA 7.12.2)
                    int ka = ((e < 8) ? 0 : 16) + khf * 8 + (e & 7);
                    a[e] = swt[aRow * 32 + ka];
                }
#pragma unroll
                for (int e = 0; e < 16; ++e) {          // B 32x16 f16 layout
                    int kb = k0 + ((e < 8) ? 0 : 16) + khf * 8 + (e & 7);
                    float v = (inb && kb < Cin) ? xt[(size_t)kb * cstr] : 0.f;
                    b[e] = (_Float16)v;
                }
                acc = __builtin_amdgcn_wmma_f32_16x16x32_f16(
                    false, a, false, b, (short)0, acc, false, false);
                __syncthreads();                        // before next chunk overwrite
            }
        }
    }

#pragma unroll
    for (int r = 0; r < 8; ++r) {                       // D layout: M = r + 8*khf, N = l15
        int mo = co0 + r + khf * 8;
        int po = p0 + l15;
        if (mo < Cout && po < HWo) {
            float v = acc[r];
            if (bng) {
                float sc = bng[mo] * rsqrtf(bnv[mo] + 1e-5f);
                v = v * sc + (bnb[mo] - bnm[mo] * sc);
            }
            size_t idx = ((size_t)n * Cout + mo) * HWo + po;
            if (res) v += res[idx];
            if (act == 1) v = fmaxf(v, 0.f);
            y[idx] = v;
        }
    }
}

// ---------------------------------------------------------------------------
// Direct conv (for Cin=1 stems and tiny 1x1 saliency head). act: 0 none,
// 1 relu, 2 sigmoid. Optional BN.
// ---------------------------------------------------------------------------
__global__ void k_conv_direct(const float* __restrict__ x, const float* __restrict__ w,
                              float* __restrict__ y,
                              const float* bng, const float* bnb,
                              const float* bnm, const float* bnv, int act,
                              int N, int Cin, int Hin, int Win,
                              int Cout, int Hout, int Wout,
                              int ks, int stride, int pad) {
    size_t total = (size_t)N * Cout * Hout * Wout;
    size_t i = (size_t)blockIdx.x * 256 + threadIdx.x;
    if (i >= total) return;
    int wo = (int)(i % Wout);
    size_t t = i / Wout;
    int ho = (int)(t % Hout); t /= Hout;
    int co = (int)(t % Cout);
    int n  = (int)(t / Cout);
    float acc = 0.f;
    for (int ci = 0; ci < Cin; ++ci)
        for (int kh = 0; kh < ks; ++kh) {
            int iy = ho * stride - pad + kh;
            if (iy < 0 || iy >= Hin) continue;
            for (int kw = 0; kw < ks; ++kw) {
                int ix = wo * stride - pad + kw;
                if (ix < 0 || ix >= Win) continue;
                acc += x[(((size_t)n * Cin + ci) * Hin + iy) * Win + ix] *
                       w[(((size_t)co * Cin + ci) * ks + kh) * ks + kw];
            }
        }
    if (bng) {
        float sc = bng[co] * rsqrtf(bnv[co] + 1e-5f);
        acc = acc * sc + (bnb[co] - bnm[co] * sc);
    }
    if (act == 1) acc = fmaxf(acc, 0.f);
    else if (act == 2) acc = 1.f / (1.f + expf(-acc));
    y[i] = acc;
}

// 3x3 stride-2 maxpool --------------------------------------------------------
__global__ void k_maxpool(const float* __restrict__ x, float* __restrict__ y,
                          int N, int C, int Hin, int Win, int Hout, int Wout, int pad) {
    size_t total = (size_t)N * C * Hout * Wout;
    size_t i = (size_t)blockIdx.x * 256 + threadIdx.x;
    if (i >= total) return;
    int wo = (int)(i % Wout);
    size_t t = i / Wout;
    int ho = (int)(t % Hout); t /= Hout;
    int c  = (int)(t % C);
    int n  = (int)(t / C);
    float m = -INFINITY;
    for (int kh = 0; kh < 3; ++kh) {
        int iy = ho * 2 - pad + kh;
        if (iy < 0 || iy >= Hin) continue;
        for (int kw = 0; kw < 3; ++kw) {
            int ix = wo * 2 - pad + kw;
            if (ix < 0 || ix >= Win) continue;
            m = fmaxf(m, x[(((size_t)n * C + c) * Hin + iy) * Win + ix]);
        }
    }
    y[i] = m;
}

// BN + ReLU elementwise (pre-activation blocks) -------------------------------
__global__ void k_bn_relu(const float* __restrict__ x, float* __restrict__ y,
                          const float* g, const float* b, const float* m,
                          const float* v, size_t total, int C, int HW) {
    size_t i = (size_t)blockIdx.x * 256 + threadIdx.x;
    if (i >= total) return;
    int c = (int)((i / HW) % C);
    float sc = g[c] * rsqrtf(v[c] + 1e-5f);
    float val = x[i] * sc + (b[c] - m[c] * sc);
    y[i] = fmaxf(val, 0.f);
}

// Sum l_conv1 over its 3 (broadcast) input channels ---------------------------
__global__ void k_sum3(const float* __restrict__ w, float* __restrict__ o, int n) {
    int i = blockIdx.x * 256 + threadIdx.x;
    if (i >= n) return;                  // n = 64*49
    int oc = i / 49, t = i % 49;
    o[i] = w[(oc * 3 + 0) * 49 + t] + w[(oc * 3 + 1) * 49 + t] + w[(oc * 3 + 2) * 49 + t];
}

// ROI retrieval: one block per image. min/max normalize + sum channels, then
// 6x { 4x4 avg-pool argmax (first-max), zero window }. All in LDS.
__global__ void __launch_bounds__(256) k_roi(const float* __restrict__ sal, int* __restrict__ locs) {
    __shared__ float cur[CAMH * CAMW];
    __shared__ float rv[256];
    __shared__ int   ri[256];
    const int n = blockIdx.x, t = threadIdx.x;
    for (int i = t; i < CAMH * CAMW; i += 256) cur[i] = 0.f;
    __syncthreads();
    for (int ch = 0; ch < 2; ++ch) {
        const float* s = sal + ((size_t)n * 2 + ch) * (CAMH * CAMW);
        float mn = INFINITY, mx = -INFINITY;
        for (int i = t; i < CAMH * CAMW; i += 256) {
            float v = s[i]; mn = fminf(mn, v); mx = fmaxf(mx, v);
        }
        rv[t] = mx; __syncthreads();
        for (int off = 128; off; off >>= 1) { if (t < off) rv[t] = fmaxf(rv[t], rv[t + off]); __syncthreads(); }
        float chmx = rv[0]; __syncthreads();
        rv[t] = mn; __syncthreads();
        for (int off = 128; off; off >>= 1) { if (t < off) rv[t] = fminf(rv[t], rv[t + off]); __syncthreads(); }
        float chmn = rv[0]; __syncthreads();
        float inv = 1.f / (chmx - chmn);
        for (int i = t; i < CAMH * CAMW; i += 256) cur[i] += (s[i] - chmn) * inv;
        __syncthreads();
    }
    for (int k = 0; k < 6; ++k) {
        float best = -INFINITY; int bi = 0x7fffffff;
        for (int p = t; p < POOLH * POOLW; p += 256) {
            int r = p / POOLW, c = p % POOLW;
            float sum = 0.f;
            for (int dy = 0; dy < 4; ++dy)
                for (int dx = 0; dx < 4; ++dx)
                    sum += cur[(r + dy) * CAMW + c + dx];
            sum *= (1.f / 16.f);
            if (sum > best || (sum == best && p < bi)) { best = sum; bi = p; }
        }
        rv[t] = best; ri[t] = bi; __syncthreads();
        for (int off = 128; off; off >>= 1) {
            if (t < off) {
                if (rv[t + off] > rv[t] || (rv[t + off] == rv[t] && ri[t + off] < ri[t])) {
                    rv[t] = rv[t + off]; ri[t] = ri[t + off];
                }
            }
            __syncthreads();
        }
        int bidx = ri[0];
        int r = bidx / POOLW, c = bidx % POOLW;
        if (t == 0) { locs[(n * 6 + k) * 2 + 0] = r * 64; locs[(n * 6 + k) * 2 + 1] = c * 64; }
        __syncthreads();
        if (t < 16) { int dy = t >> 2, dx = t & 3; cur[(r + dy) * CAMW + (c + dx)] = 0.f; }
        __syncthreads();
    }
}

// Crop gather: one block per crop. TDM path: pull 256x64 slabs of the source
// image into LDS with tensor_load_to_lds (D# built on the fly), then fan out.
__global__ void __launch_bounds__(256)
k_crops2(const float* __restrict__ x, const int* __restrict__ locs,
         float* __restrict__ crops) {
    const int nk = blockIdx.x;          // 12 crops
    const int n  = nk / 6;
    const int r  = locs[nk * 2], c = locs[nk * 2 + 1];
#if USE_TDM
    __shared__ __align__(16) float tile[64 * 256];      // 64 KB slab
    for (int it = 0; it < 4; ++it) {
        if (threadIdx.x < 32) {                         // one wave issues the DMA
            unsigned long long ga = (unsigned long long)(const void*)
                (x + ((size_t)n * IMG_H + (r + it * 64)) * IMG_W + c);
            tdu4 g0;
            g0[0] = 1u;                                  // count=1 valid descriptor
            g0[1] = 0u;                                  // lds_addr (bytes)
            g0[2] = (unsigned)ga;                        // global_addr[31:0]
            g0[3] = ((unsigned)(ga >> 32) & 0x1FFFFFFu)  // global_addr[56:32]
                    | (2u << 30);                        // type=2 ("image")
            tdi8 g1;
            g1[0] = 0x00020000;                          // data_size=4B, no multicast
            g1[1] = (int)((unsigned)IMG_W << 16);        // tensor_dim0 lo16
            g1[2] = (int)((unsigned)IMG_H << 16);        // dim0 hi | tensor_dim1 lo16
            g1[3] = (int)(256u << 16);                   // dim1 hi | tile_dim0=256
            g1[4] = 64;                                  // tile_dim1=64, tile_dim2=0
            g1[5] = IMG_W;                               // tensor_dim0_stride lo32
            g1[6] = 0;                                   // stride0 hi | stride1 lo
            g1[7] = 0;
            tdi4 g2 = (tdi4)0, g3 = (tdi4)0;             // 2-D tensor: unused groups
            tdi8 g4 = (tdi8)0;                           // 6-arg form: extra group
            __builtin_amdgcn_tensor_load_to_lds(g0, g1, g2, g3, g4, 0);
            __builtin_amdgcn_s_wait_tensorcnt(0);
        }
        __syncthreads();
        float* dst = crops + (size_t)nk * 65536 + (size_t)it * 64 * 256;
        for (int i = threadIdx.x; i < 64 * 256; i += 256) dst[i] = tile[i];
        __syncthreads();
    }
#else
    for (int i = threadIdx.x; i < 65536; i += 256) {
        int dy = i >> 8, dx = i & 255;
        crops[(size_t)nk * 65536 + i] =
            x[((size_t)n * IMG_H + r + dy) * IMG_W + c + dx];
    }
#endif
}

// Global average over HW (local-net head) -------------------------------------
__global__ void k_avg(const float* __restrict__ x, float* __restrict__ o, int HW, int total) {
    int i = blockIdx.x * 256 + threadIdx.x;
    if (i >= total) return;
    float acc = 0.f;
    const float* p = x + (size_t)i * HW;
    for (int j = 0; j < HW; ++j) acc += p[j];
    o[i] = acc / (float)HW;
}

// Max over HW (global feature g) ----------------------------------------------
__global__ void k_maxsp(const float* __restrict__ x, float* __restrict__ o, int HW, int total) {
    int i = blockIdx.x * 256 + threadIdx.x;
    if (i >= total) return;
    float m = -INFINITY;
    const float* p = x + (size_t)i * HW;
    for (int j = 0; j < HW; ++j) m = fmaxf(m, p[j]);
    o[i] = m;
}

// Gated attention scores: block per (n,k); 128 threads = gate dims ------------
__global__ void __launch_bounds__(128)
k_att(const float* __restrict__ hc, const float* __restrict__ V,
      const float* __restrict__ U, const float* __restrict__ w, float* __restrict__ s) {
    __shared__ float red[128];
    const float* h = hc + (size_t)blockIdx.x * 512;
    int j = threadIdx.x;
    float dv = 0.f, du = 0.f;
    for (int i = 0; i < 512; ++i) {
        float hv = h[i];
        dv += hv * V[j * 512 + i];
        du += hv * U[j * 512 + i];
    }
    float gate = tanhf(dv) * (1.f / (1.f + expf(-du)));
    red[j] = gate * w[j];
    __syncthreads();
    for (int off = 64; off; off >>= 1) { if (j < off) red[j] += red[j + off]; __syncthreads(); }
    if (j == 0) s[blockIdx.x] = red[0];
}

__global__ void k_alpha(const float* __restrict__ s, float* __restrict__ alpha) {
    int t = threadIdx.x;
    if (t < 2) {
        float mx = -INFINITY;
        for (int k = 0; k < 6; ++k) mx = fmaxf(mx, s[t * 6 + k]);
        float e[6], sum = 0.f;
        for (int k = 0; k < 6; ++k) { e[k] = expf(s[t * 6 + k] - mx); sum += e[k]; }
        for (int k = 0; k < 6; ++k) alpha[t * 6 + k] = e[k] / sum;
    }
}

__global__ void k_z(const float* __restrict__ alpha, const float* __restrict__ hc,
                    float* __restrict__ z) {
    int i = blockIdx.x * 256 + threadIdx.x;
    if (i >= 1024) return;
    int n = i >> 9, d = i & 511;
    float acc = 0.f;
    for (int k = 0; k < 6; ++k) acc += alpha[n * 6 + k] * hc[((size_t)(n * 6 + k) << 9) + d];
    z[i] = acc;
}

__global__ void k_final(const float* __restrict__ g, const float* __restrict__ z,
                        const float* __restrict__ W, const float* __restrict__ b,
                        float* __restrict__ out) {
    int i = threadIdx.x;
    if (i >= 4) return;
    int n = i >> 1, c = i & 1;
    float acc = b[c];
    for (int j = 0; j < 256; ++j) acc += g[n * 256 + j] * W[c * 768 + j];
    for (int j = 0; j < 512; ++j) acc += z[n * 512 + j] * W[c * 768 + 256 + j];
    out[i] = 1.f / (1.f + expf(-acc));
}

// ===========================================================================
// Host orchestration
// ===========================================================================
struct BNp { const float *g, *b, *m, *v; };

static inline int cdiv_(long long a, long long b) { return (int)((a + b - 1) / b); }
static inline int ceil32_(int x) { return (x + 31) & ~31; }

extern "C" void kernel_launch(void* const* d_in, const int* in_sizes, int n_in,
                              void* d_out, int out_size, void* d_ws, size_t ws_size,
                              hipStream_t stream) {
    (void)in_sizes; (void)n_in; (void)out_size; (void)ws_size;

    // ---- walk input leaves (dict insertion order, nested params sorted keys)
    int ic = 0;
    auto nf  = [&]() { return (const float*)d_in[ic++]; };
    auto nbn = [&](BNp& bn) { bn.g = nf(); bn.b = nf(); bn.m = nf(); bn.v = nf(); };

    const float* X    = nf();          // x
    const float* attU = nf();
    const float* attV = nf();
    nf();                              // att_cls (unused)
    const float* attw = nf();
    const float* fusW = nf();
    const float* fusb = nf();
    BNp gbnf; nbn(gbnf);
    const float* gconv1 = nf();
    BNp gbn1[5][2], gbn2[5][2];
    const float *gc1[5][2], *gc2[5][2], *gdn[5][2];
    for (int li = 0; li < 5; ++li)
        for (int b = 0; b < 2; ++b) {
            nbn(gbn1[li][b]); nbn(gbn2[li][b]);
            gc1[li][b] = nf(); gc2[li][b] = nf();
            gdn[li][b] = (li > 0 && b == 0) ? nf() : nullptr;
        }
    BNp lbn1s; nbn(lbn1s);
    const float* lconv1 = nf();
    BNp lbn1[4][2], lbn2[4][2], ldbn[4][2];
    const float *lc1[4][2], *lc2[4][2], *ldn[4][2];
    for (int li = 0; li < 4; ++li)
        for (int b = 0; b < 2; ++b) {
            nbn(lbn1[li][b]); nbn(lbn2[li][b]);
            lc1[li][b] = nf(); lc2[li][b] = nf();
            if (li > 0 && b == 0) { ldn[li][b] = nf(); nbn(ldbn[li][b]); }
            else ldn[li][b] = nullptr;
        }
    const float* salw = nf();

    // ---- workspace layout
    size_t wcur = 0;
    auto walloc = [&](size_t bytes) -> void* {
        void* p = (void*)((char*)d_ws + wcur);
        wcur += (bytes + 255) & ~(size_t)255;
        return p;
    };
    float* bufA   = (float*)walloc((size_t)45219840 * 4);  // stem out, reused
    float* bufH   = (float*)walloc((size_t)11266080 * 4);
    float* bufH2  = (float*)walloc((size_t)11266080 * 4);
    float* bufO   = (float*)walloc((size_t)11266080 * 4);
    float* bufT   = bufA;                                  // <=12M floats
    float* bufR   = bufA + 12000000;                       // <=12M floats
    float* bufLc1 = bufA + 24000000;                       // 12.58M floats
    float* sSal   = (float*)walloc(5520 * 4);
    float* sG     = (float*)walloc(512 * 4);
    int*   sLocs  = (int*)walloc(32 * 4);
    float* sCrops = (float*)walloc((size_t)786432 * 4);
    float* wSum   = (float*)walloc(3136 * 4);
    float* sHC    = (float*)walloc(6144 * 4);
    float* sSC    = (float*)walloc(16 * 4);
    float* sAL    = (float*)walloc(16 * 4);
    float* sZ     = (float*)walloc(1024 * 4);

    // ---- pack all WMMA weights to f16 up front
    auto pack = [&](const float* w, int cout, int cin, int ks) -> const _Float16* {
        int cop = ceil32_(cout), cip = ceil32_(cin);
        size_t n = (size_t)ks * ks * cop * cip;
        _Float16* p = (_Float16*)walloc(n * sizeof(_Float16));
        k_pack_w<<<cdiv_(n, 256), 256, 0, stream>>>(w, p, cout, cin, ks, cop, cip);
        return p;
    };
    const _Float16 *pkGc1[5][2], *pkGc2[5][2], *pkGdn[5][2];
    {
        int inp = 16;
        for (int li = 0; li < 5; ++li) {
            int planes = 16 << li;
            for (int b = 0; b < 2; ++b) {
                pkGc1[li][b] = pack(gc1[li][b], planes, inp, 3);
                pkGc2[li][b] = pack(gc2[li][b], planes, planes, 3);
                pkGdn[li][b] = gdn[li][b] ? pack(gdn[li][b], planes, inp, 1) : nullptr;
                inp = planes;
            }
        }
    }
    const _Float16 *pkLc1[4][2], *pkLc2[4][2], *pkLdn[4][2];
    {
        int inp = 64;
        for (int li = 0; li < 4; ++li) {
            int planes = 64 << li;
            for (int b = 0; b < 2; ++b) {
                pkLc1[li][b] = pack(lc1[li][b], planes, inp, 3);
                pkLc2[li][b] = pack(lc2[li][b], planes, planes, 3);
                pkLdn[li][b] = ldn[li][b] ? pack(ldn[li][b], planes, inp, 1) : nullptr;
                inp = planes;
            }
        }
    }

    auto convW = [&](const float* x, const _Float16* wpk, float* y, const float* res,
                     const BNp* bn, int act, int n, int cin, int hin, int win,
                     int cout, int ks, int st, int pad, int ho, int wo) {
        int cop = ceil32_(cout);
        dim3 grid(cdiv_((long long)ho * wo, 64), n * (cop >> 5));
        k_conv_wmma<<<grid, 256, 0, stream>>>(
            x, wpk, y, res,
            bn ? bn->g : nullptr, bn ? bn->b : nullptr,
            bn ? bn->m : nullptr, bn ? bn->v : nullptr,
            n, cin, hin, win, cout, ho, wo, ks, st, pad,
            ceil32_(cin), cop, act);
    };

    // ================= Global backbone =================
    // conv1 7x7 s2 p3 : (2,1,2944,1920) -> (2,16,1472,960)
    k_conv_direct<<<cdiv_((long long)2 * 16 * 1472 * 960, 256), 256, 0, stream>>>(
        X, gconv1, bufA, nullptr, nullptr, nullptr, nullptr, 0,
        2, 1, IMG_H, IMG_W, 16, 1472, 960, 7, 2, 3);
    // maxpool 3x3 s2 p0 -> (2,16,735,479)
    k_maxpool<<<cdiv_((long long)2 * 16 * 735 * 479, 256), 256, 0, stream>>>(
        bufA, bufH, 2, 16, 1472, 960, 735, 479, 0);

    int Hs = 735, Ws = 479, inp = 16;
    float *h = bufH, *h2 = bufH2;
    const int GST[5] = {1, 2, 2, 2, 2};
    for (int li = 0; li < 5; ++li) {
        int planes = 16 << li;
        for (int b = 0; b < 2; ++b) {
            int st = (b == 0) ? GST[li] : 1;
            int Ho = (Hs + 2 - 3) / st + 1;
            int Wo = (Ws + 2 - 3) / st + 1;
            size_t tot = (size_t)2 * inp * Hs * Ws;
            k_bn_relu<<<cdiv_(tot, 256), 256, 0, stream>>>(
                h, bufO, gbn1[li][b].g, gbn1[li][b].b, gbn1[li][b].m, gbn1[li][b].v,
                tot, inp, Hs * Ws);
            const float* resp;
            if (pkGdn[li][b]) {
                convW(bufO, pkGdn[li][b], bufR, nullptr, nullptr, 0,
                      2, inp, Hs, Ws, planes, 1, st, 0, Ho, Wo);
                resp = bufR;
            } else resp = h;
            convW(bufO, pkGc1[li][b], bufT, nullptr, nullptr, 0,
                  2, inp, Hs, Ws, planes, 3, st, 1, Ho, Wo);
            size_t tot2 = (size_t)2 * planes * Ho * Wo;
            k_bn_relu<<<cdiv_(tot2, 256), 256, 0, stream>>>(
                bufT, bufO, gbn2[li][b].g, gbn2[li][b].b, gbn2[li][b].m, gbn2[li][b].v,
                tot2, planes, Ho * Wo);
            convW(bufO, pkGc2[li][b], h2, resp, nullptr, 0,
                  2, planes, Ho, Wo, planes, 3, 1, 1, Ho, Wo);
            float* tmp = h; h = h2; h2 = tmp;
            Hs = Ho; Ws = Wo; inp = planes;
        }
    }
    // h_g = relu(bnf(h)) : (2,256,46,30)
    {
        size_t tot = (size_t)2 * 256 * CAMH * CAMW;
        k_bn_relu<<<cdiv_(tot, 256), 256, 0, stream>>>(
            h, bufO, gbnf.g, gbnf.b, gbnf.m, gbnf.v, tot, 256, CAMH * CAMW);
    }
    const float* hg = bufO;
    // saliency 1x1 + sigmoid
    k_conv_direct<<<cdiv_((long long)2 * 2 * CAMH * CAMW, 256), 256, 0, stream>>>(
        hg, salw, sSal, nullptr, nullptr, nullptr, nullptr, 2,
        2, 256, CAMH, CAMW, 2, CAMH, CAMW, 1, 1, 0);
    // g = max over spatial
    k_maxsp<<<cdiv_(512, 256), 256, 0, stream>>>(hg, sG, CAMH * CAMW, 512);
    // ROI + crops (crops use the Tensor Data Mover when available)
    k_roi<<<2, 256, 0, stream>>>(sSal, sLocs);
    k_crops2<<<12, 256, 0, stream>>>(X, sLocs, sCrops);

    // ================= Local network =================
    k_sum3<<<cdiv_(3136, 256), 256, 0, stream>>>(lconv1, wSum, 3136);
    // conv1 7x7 s2 p3 + bn + relu : (12,1,256,256) -> (12,64,128,128)
    k_conv_direct<<<cdiv_((long long)12 * 64 * 128 * 128, 256), 256, 0, stream>>>(
        sCrops, wSum, bufLc1, lbn1s.g, lbn1s.b, lbn1s.m, lbn1s.v, 1,
        12, 1, 256, 256, 64, 128, 128, 7, 2, 3);
    // maxpool 3x3 s2 p1 -> (12,64,64,64)
    k_maxpool<<<cdiv_((long long)12 * 64 * 64 * 64, 256), 256, 0, stream>>>(
        bufLc1, bufH, 12, 64, 128, 128, 64, 64, 1);

    Hs = 64; Ws = 64; inp = 64;
    h = bufH; h2 = bufH2;
    const int LST[4] = {1, 2, 2, 2};
    for (int li = 0; li < 4; ++li) {
        int planes = 64 << li;
        for (int b = 0; b < 2; ++b) {
            int st = (b == 0) ? LST[li] : 1;
            int Ho = (Hs + 2 - 3) / st + 1;
            int Wo = (Ws + 2 - 3) / st + 1;
            convW(h, pkLc1[li][b], bufT, nullptr, &lbn1[li][b], 1,
                  12, inp, Hs, Ws, planes, 3, st, 1, Ho, Wo);
            const float* resp;
            if (pkLdn[li][b]) {
                convW(h, pkLdn[li][b], bufR, nullptr, &ldbn[li][b], 0,
                      12, inp, Hs, Ws, planes, 1, st, 0, Ho, Wo);
                resp = bufR;
            } else resp = h;
            convW(bufT, pkLc2[li][b], h2, resp, &lbn2[li][b], 1,
                  12, planes, Ho, Wo, planes, 3, 1, 1, Ho, Wo);
            float* tmp = h; h = h2; h2 = tmp;
            Hs = Ho; Ws = Wo; inp = planes;
        }
    }
    // hc = mean over 8x8 : (12,512)
    k_avg<<<cdiv_(6144, 256), 256, 0, stream>>>(h, sHC, 64, 6144);

    // ================= Attention + fusion =================
    k_att<<<12, 128, 0, stream>>>(sHC, attV, attU, attw, sSC);
    k_alpha<<<1, 32, 0, stream>>>(sSC, sAL);
    k_z<<<4, 256, 0, stream>>>(sAL, sHC, sZ);
    k_final<<<1, 32, 0, stream>>>(sG, sZ, fusW, fusb, (float*)d_out);
}
